// PegSolitaireGNN_36524401885972
// MI455X (gfx1250) — matrix operand order repaired and checked by reading, not compiled, not run on hardware
//
#include <hip/hip_runtime.h>
#include <hip/hip_bf16.h>

// ---------------------------------------------------------------------------
// PegSolitaire GATv2 GNN for gfx1250 (MI455X).
// Dense node transforms (layers 2/3) use v_wmma_f32_16x16x32_bf16.
// Edge softmax/aggregation uses float atomics (gather/scatter bound).
// ---------------------------------------------------------------------------

#define HEADS 4
#define CH    128
#define NEG_SLOPE 0.2f

typedef __attribute__((ext_vector_type(16))) __bf16 v16bf;
typedef __attribute__((ext_vector_type(8)))  float  v8f;
typedef __attribute__((ext_vector_type(4)))  unsigned int u32x4;

union Frag32B { v16bf v; u32x4 q[2]; };

// float -> bf16 (round-to-nearest-even)
__device__ __forceinline__ unsigned short f2bf(float f) {
    unsigned u = __float_as_uint(f);
    unsigned r = (u + 0x7FFFu + ((u >> 16) & 1u)) >> 16;
    return (unsigned short)r;
}

// order-preserving float<->uint encoding for atomicMax
__device__ __forceinline__ unsigned fenc(float f) {
    unsigned u = __float_as_uint(f);
    return (u & 0x80000000u) ? ~u : (u | 0x80000000u);
}
__device__ __forceinline__ float fdec(unsigned e) {
    unsigned u = (e & 0x80000000u) ? (e ^ 0x80000000u) : ~e;
    return __uint_as_float(u);
}

// ---------------------------------------------------------------------------
// generic fill
__global__ void k_fill_u32(unsigned* p, unsigned val, long long n) {
    long long i = (long long)blockIdx.x * blockDim.x + threadIdx.x;
    if (i < n) p[i] = val;
}

// ---------------------------------------------------------------------------
// layer-1 tiny-K GEMM: out[n,c] = b[c] + sum_{k<4} x[n,k]*W[k,c]
__global__ void k_gemm_k4(const float* __restrict__ x, const float* __restrict__ W,
                          const float* __restrict__ b, float* __restrict__ out,
                          int nrows, int OUT) {
    long long i = (long long)blockIdx.x * blockDim.x + threadIdx.x;
    if (i >= (long long)nrows * OUT) return;
    int n = (int)(i / OUT), c = (int)(i % OUT);
    const float* xp = x + (long long)n * 4;
    float acc = b[c];
    acc += xp[0] * W[0 * OUT + c];
    acc += xp[1] * W[1 * OUT + c];
    acc += xp[2] * W[2 * OUT + c];
    acc += xp[3] * W[3 * OUT + c];
    out[i] = acc;
}

// ---------------------------------------------------------------------------
// convert + transpose weights:  W[K,N] f32  ->  Wt[N,K] bf16
__global__ void k_convert_transpose(const float* __restrict__ W, unsigned short* __restrict__ Wt,
                                    int K, int N) {
    long long i = (long long)blockIdx.x * blockDim.x + threadIdx.x;
    if (i >= (long long)K * N) return;
    int k = (int)(i / N), n = (int)(i % N);
    Wt[(long long)n * K + k] = f2bf(W[i]);
}

// ---------------------------------------------------------------------------
// WMMA GEMM: C[M,N] = A[M,K](bf16, row-major) * Bt[N,K](bf16)^T + bias[N]
// One wave computes a 16x64 strip: 4 accumulators share one A fragment.
// K multiple of 32, M multiple of 16, N multiple of 64 assumed.
__global__ void k_wmma_gemm_bias(const unsigned short* __restrict__ A,
                                 const unsigned short* __restrict__ Bt,
                                 const float* __restrict__ bias,
                                 float* __restrict__ C,
                                 int Mtiles, int NG, int N, int K) {
    int wave = (int)((blockIdx.x * blockDim.x + threadIdx.x) >> 5);
    if (wave >= Mtiles * NG) return;            // wave-uniform: EXEC stays full
    int lane = threadIdx.x & 31;
    int mt = wave / NG;
    int ng = wave % NG;

    int arow  = mt * 16 + (lane & 15);
    int kseg  = (lane >> 4) * 16;                // A: half-wave K sub-block
    int bksel = (lane >> 4) * 16;                // B: half-wave K sub-block
    int colbase = ng * 64;
    int bcol = lane & 15;

    v8f acc0 = {}, acc1 = {}, acc2 = {}, acc3 = {};
    const unsigned short* arowp = A + (long long)arow * K;

    for (int k0 = 0; k0 < K; k0 += 32) {
        Frag32B a;
        const unsigned short* ap = arowp + k0 + kseg;
        a.q[0] = *(const u32x4*)(ap);            // K = k0+kseg .. +7
        a.q[1] = *(const u32x4*)(ap + 8);        // K = k0+kseg+8 .. +15

        Frag32B b0, b1, b2, b3;
        const unsigned short* bp0 = Bt + (long long)(colbase +  0 + bcol) * K + k0 + bksel;
        const unsigned short* bp1 = Bt + (long long)(colbase + 16 + bcol) * K + k0 + bksel;
        const unsigned short* bp2 = Bt + (long long)(colbase + 32 + bcol) * K + k0 + bksel;
        const unsigned short* bp3 = Bt + (long long)(colbase + 48 + bcol) * K + k0 + bksel;
        b0.q[0] = *(const u32x4*)(bp0); b0.q[1] = *(const u32x4*)(bp0 + 8);
        b1.q[0] = *(const u32x4*)(bp1); b1.q[1] = *(const u32x4*)(bp1 + 8);
        b2.q[0] = *(const u32x4*)(bp2); b2.q[1] = *(const u32x4*)(bp2 + 8);
        b3.q[0] = *(const u32x4*)(bp3); b3.q[1] = *(const u32x4*)(bp3 + 8);

        acc0 = __builtin_amdgcn_wmma_f32_16x16x32_bf16(false, a.v, false, b0.v, (short)0, acc0, false, false);
        acc1 = __builtin_amdgcn_wmma_f32_16x16x32_bf16(false, a.v, false, b1.v, (short)0, acc1, false, false);
        acc2 = __builtin_amdgcn_wmma_f32_16x16x32_bf16(false, a.v, false, b2.v, (short)0, acc2, false, false);
        acc3 = __builtin_amdgcn_wmma_f32_16x16x32_bf16(false, a.v, false, b3.v, (short)0, acc3, false, false);
    }

    // store: element v of acc -> row mt*16 + v + 8*(lane>>4), col colbase + t*16 + (lane&15)
    int rbase = mt * 16 + 8 * (lane >> 4);
    v8f accs[4] = {acc0, acc1, acc2, acc3};
#pragma unroll
    for (int t = 0; t < 4; ++t) {
        int col = colbase + t * 16 + bcol;
        float bv = bias[col];
#pragma unroll
        for (int v = 0; v < 8; ++v) {
            C[(long long)(rbase + v) * N + col] = accs[t][v] + bv;
        }
    }
}

// ---------------------------------------------------------------------------
// edge pass 1: score[e,h] = att[h,:] . leaky_relu(xl[src]+xr[dst]); atomicMax per dst
// one wave per (edge, head)
__global__ void k_edge_score(const float* __restrict__ xl, const float* __restrict__ xr,
                             const float* __restrict__ att, const int* __restrict__ ei,
                             int E, int Etot, int H,
                             float* __restrict__ score, unsigned* __restrict__ nmax) {
    int w = (int)((blockIdx.x * blockDim.x + threadIdx.x) >> 5);
    if (w >= Etot * H) return;
    int lane = threadIdx.x & 31;
    int e = w / H, h = w % H;
    int src, dst;
    if (e < E) { src = ei[e]; dst = ei[E + e]; } else { src = dst = e - E; }
    int HC = H * CH;
    const float* pl = xl + (long long)src * HC + h * CH;
    const float* pr = xr + (long long)dst * HC + h * CH;
    const float* pa = att + h * CH;
    float s = 0.f;
#pragma unroll
    for (int i = 0; i < 4; ++i) {
        int c = lane + 32 * i;
        float v = pl[c] + pr[c];
        v = (v > 0.f) ? v : NEG_SLOPE * v;
        s += v * pa[c];
    }
#pragma unroll
    for (int off = 16; off > 0; off >>= 1) s += __shfl_xor(s, off, 32);
    if (lane == 0) {
        score[(long long)e * H + h] = s;
        atomicMax(&nmax[dst * H + h], fenc(s));
    }
}

// edge pass 2: ex = exp(score - max[dst]); atomicAdd denominator
__global__ void k_edge_exp(const int* __restrict__ ei, int E, int Etot, int H,
                           float* __restrict__ score, const unsigned* __restrict__ nmax,
                           float* __restrict__ den) {
    int i = (int)(blockIdx.x * blockDim.x + threadIdx.x);
    if (i >= Etot * H) return;
    int e = i / H, h = i % H;
    int dst = (e < E) ? ei[E + e] : (e - E);
    float m = fdec(nmax[dst * H + h]);
    float ex = __expf(score[i] - m);
    score[i] = ex;
    atomicAdd(&den[dst * H + h], ex);
}

// edge pass 3: out[dst] += alpha * xl[src];  one wave per (edge, head)
__global__ void k_edge_msg(const float* __restrict__ xl, const int* __restrict__ ei,
                           int E, int Etot, int H,
                           const float* __restrict__ score, const float* __restrict__ den,
                           float* __restrict__ out) {
    int w = (int)((blockIdx.x * blockDim.x + threadIdx.x) >> 5);
    if (w >= Etot * H) return;
    int lane = threadIdx.x & 31;
    int e = w / H, h = w % H;
    int src, dst;
    if (e < E) { src = ei[e]; dst = ei[E + e]; } else { src = dst = e - E; }
    int HC = H * CH;
    float alpha = score[(long long)e * H + h] / den[dst * H + h];
    const float* pl = xl + (long long)src * HC + h * CH;
    float* po = out + (long long)dst * HC + h * CH;
#pragma unroll
    for (int i = 0; i < 4; ++i) {
        int c = lane + 32 * i;
        atomicAdd(&po[c], pl[c] * alpha);
    }
}

// finalize: h = elu(out + bias); also emit bf16 copy for next layer's WMMA
__global__ void k_finalize_elu(float* __restrict__ out, const float* __restrict__ bias,
                               unsigned short* __restrict__ hbf, int nrows, int HC) {
    long long i = (long long)blockIdx.x * blockDim.x + threadIdx.x;
    if (i >= (long long)nrows * HC) return;
    int c = (int)(i % HC);
    float v = out[i] + bias[c];
    v = (v > 0.f) ? v : (__expf(v) - 1.f);
    out[i] = v;
    if (hbf) hbf[i] = f2bf(v);
}

// ---------------------------------------------------------------------------
// pooling
__global__ void k_pool_count(const int* __restrict__ batch, float* __restrict__ cnt, int nrows) {
    int n = (int)(blockIdx.x * blockDim.x + threadIdx.x);
    if (n >= nrows) return;
    atomicAdd(&cnt[batch[n]], 1.0f);
}
__global__ void k_pool_sum(const float* __restrict__ h, const int* __restrict__ batch,
                           float* __restrict__ gsum, int nrows) {
    long long i = (long long)blockIdx.x * blockDim.x + threadIdx.x;
    if (i >= (long long)nrows * CH) return;
    int n = (int)(i / CH), c = (int)(i % CH);
    atomicAdd(&gsum[batch[n] * CH + c], h[i]);
}
__global__ void k_build_comb(const float* __restrict__ gsum, const float* __restrict__ cnt,
                             const float* __restrict__ pegs, const float* __restrict__ moves,
                             float* __restrict__ comb, int G) {
    int i = (int)(blockIdx.x * blockDim.x + threadIdx.x);
    if (i >= G * (CH + 2)) return;
    int g = i / (CH + 2), c = i % (CH + 2);
    float v;
    if (c < CH)            v = gsum[g * CH + c] / fmaxf(cnt[g], 1.0f);
    else if (c == CH)      v = pegs[g];
    else                   v = moves[g];
    comb[i] = v;
}

// small MLPs for heads
__global__ void k_mlp_relu(const float* __restrict__ in, const float* __restrict__ W,
                           const float* __restrict__ b, float* __restrict__ out,
                           int G, int K, int OUT) {
    int i = (int)(blockIdx.x * blockDim.x + threadIdx.x);
    if (i >= G * OUT) return;
    int g = i / OUT, j = i % OUT;
    float acc = b[j];
    for (int k = 0; k < K; ++k) acc += in[g * K + k] * W[k * OUT + j];
    out[i] = fmaxf(acc, 0.f);
}
__global__ void k_head_policy(const float* __restrict__ hid, const float* __restrict__ W,
                              const float* __restrict__ b, float* __restrict__ out, int G) {
    int i = (int)(blockIdx.x * blockDim.x + threadIdx.x);
    if (i >= G * CH) return;
    int g = i / CH, a = i % CH;
    float acc = b[a];
    for (int j = 0; j < 128; ++j) acc += hid[g * 128 + j] * W[j * CH + a];
    out[i] = acc;
}
__global__ void k_head_value(const float* __restrict__ hid, const float* __restrict__ W,
                             const float* __restrict__ b, float* __restrict__ out, int G) {
    int g = (int)(blockIdx.x * blockDim.x + threadIdx.x);
    if (g >= G) return;
    float acc = b[0];
    for (int j = 0; j < 128; ++j) acc += hid[g * 128 + j] * W[j];
    out[g] = tanhf(acc);
}

// ---------------------------------------------------------------------------
extern "C" void kernel_launch(void* const* d_in, const int* in_sizes, int n_in,
                              void* d_out, int out_size, void* d_ws, size_t ws_size,
                              hipStream_t stream) {
    const float* x          = (const float*)d_in[0];
    const int*   edge_index = (const int*)  d_in[1];
    const int*   batch      = (const int*)  d_in[2];
    const float* pegs_left  = (const float*)d_in[3];
    const float* move_count = (const float*)d_in[4];
    const float* W1l = (const float*)d_in[5];  const float* b1l = (const float*)d_in[6];
    const float* W1r = (const float*)d_in[7];  const float* b1r = (const float*)d_in[8];
    const float* att1 = (const float*)d_in[9]; const float* bias1 = (const float*)d_in[10];
    const float* W2l = (const float*)d_in[11]; const float* b2l = (const float*)d_in[12];
    const float* W2r = (const float*)d_in[13]; const float* b2r = (const float*)d_in[14];
    const float* att2 = (const float*)d_in[15]; const float* bias2 = (const float*)d_in[16];
    const float* W3l = (const float*)d_in[17]; const float* b3l = (const float*)d_in[18];
    const float* W3r = (const float*)d_in[19]; const float* b3r = (const float*)d_in[20];
    const float* att3 = (const float*)d_in[21]; const float* bias3 = (const float*)d_in[22];
    const float* vW1 = (const float*)d_in[23]; const float* vb1 = (const float*)d_in[24];
    const float* vW2 = (const float*)d_in[25]; const float* vb2 = (const float*)d_in[26];
    const float* pW1 = (const float*)d_in[27]; const float* pb1 = (const float*)d_in[28];
    const float* pW2 = (const float*)d_in[29]; const float* pb2 = (const float*)d_in[30];

    const int N    = in_sizes[0] / 4;       // 20000
    const int E    = in_sizes[1] / 2;       // 160000
    const int G    = in_sizes[3];           // 64
    const int Etot = E + N;                 // self-loops appended
    const int HC   = HEADS * CH;            // 512

    // bump allocator over workspace
    char* base = (char*)d_ws;
    auto alloc = [&](size_t bytes) -> void* {
        void* p = (void*)base;
        base += (bytes + 255) & ~(size_t)255;
        return p;
    };
    float*          xl    = (float*)alloc((size_t)N * HC * 4);
    float*          xr    = (float*)alloc((size_t)N * HC * 4);
    float*          hbuf  = (float*)alloc((size_t)N * HC * 4);
    unsigned short* hbf   = (unsigned short*)alloc((size_t)N * HC * 2);
    float*          score = (float*)alloc((size_t)Etot * HEADS * 4);
    float*          den   = (float*)alloc((size_t)N * HEADS * 4);
    unsigned*       nmax  = (unsigned*)alloc((size_t)N * HEADS * 4);
    unsigned short* wbt1  = (unsigned short*)alloc((size_t)HC * HC * 2);
    unsigned short* wbt2  = (unsigned short*)alloc((size_t)HC * HC * 2);
    float*          gsum  = (float*)alloc((size_t)G * CH * 4);
    float*          cnt   = (float*)alloc((size_t)G * 4);
    float*          comb  = (float*)alloc((size_t)G * (CH + 2) * 4);
    float*          hidv  = (float*)alloc((size_t)G * 128 * 4);
    float*          hidp  = (float*)alloc((size_t)G * 128 * 4);

    const int TB = 256;
    auto blocks1 = [&](long long n) { return (unsigned)((n + TB - 1) / TB); };

    // ---- one GATv2 attention/aggregation round (given xl, xr already computed) ----
    auto gat_edges = [&](const float* att, const float* bias, int H, unsigned short* hbf_out) {
        long long nhc = (long long)N * H * CH;
        k_fill_u32<<<blocks1((long long)N * H), TB, 0, stream>>>(nmax, 0u, (long long)N * H);
        k_fill_u32<<<blocks1((long long)N * H), TB, 0, stream>>>((unsigned*)den, 0u, (long long)N * H);
        k_fill_u32<<<blocks1(nhc), TB, 0, stream>>>((unsigned*)hbuf, 0u, nhc);
        long long waves = (long long)Etot * H;
        k_edge_score<<<blocks1(waves * 32), TB, 0, stream>>>(xl, xr, att, edge_index, E, Etot, H, score, nmax);
        k_edge_exp<<<blocks1(waves), TB, 0, stream>>>(edge_index, E, Etot, H, score, nmax, den);
        k_edge_msg<<<blocks1(waves * 32), TB, 0, stream>>>(xl, edge_index, E, Etot, H, score, den, hbuf);
        k_finalize_elu<<<blocks1(nhc), TB, 0, stream>>>(hbuf, bias, hbf_out, N, H * CH);
    };

    // ================= layer 1 (K=4, scalar GEMM) =================
    k_gemm_k4<<<blocks1((long long)N * HC), TB, 0, stream>>>(x, W1l, b1l, xl, N, HC);
    k_gemm_k4<<<blocks1((long long)N * HC), TB, 0, stream>>>(x, W1r, b1r, xr, N, HC);
    gat_edges(att1, bias1, HEADS, hbf);

    // ================= layer 2 (WMMA bf16 GEMM, 512x512) =================
    k_convert_transpose<<<blocks1((long long)HC * HC), TB, 0, stream>>>(W2l, wbt1, HC, HC);
    k_convert_transpose<<<blocks1((long long)HC * HC), TB, 0, stream>>>(W2r, wbt2, HC, HC);
    {
        int Mtiles = N / 16, NG = HC / 64;
        long long waves = (long long)Mtiles * NG;
        k_wmma_gemm_bias<<<blocks1(waves * 32), TB, 0, stream>>>(hbf, wbt1, b2l, xl, Mtiles, NG, HC, HC);
        k_wmma_gemm_bias<<<blocks1(waves * 32), TB, 0, stream>>>(hbf, wbt2, b2r, xr, Mtiles, NG, HC, HC);
    }
    gat_edges(att2, bias2, HEADS, hbf);

    // ================= layer 3 (WMMA bf16 GEMM, 512x128, heads=1) =================
    k_convert_transpose<<<blocks1((long long)HC * CH), TB, 0, stream>>>(W3l, wbt1, HC, CH);
    k_convert_transpose<<<blocks1((long long)HC * CH), TB, 0, stream>>>(W3r, wbt2, HC, CH);
    {
        int Mtiles = N / 16, NG = CH / 64;
        long long waves = (long long)Mtiles * NG;
        k_wmma_gemm_bias<<<blocks1(waves * 32), TB, 0, stream>>>(hbf, wbt1, b3l, xl, Mtiles, NG, CH, HC);
        k_wmma_gemm_bias<<<blocks1(waves * 32), TB, 0, stream>>>(hbf, wbt2, b3r, xr, Mtiles, NG, CH, HC);
    }
    gat_edges(att3, bias3, 1, nullptr);   // result (elu'd) lives in hbuf [N,128]

    // ================= global mean pool + heads =================
    k_fill_u32<<<blocks1((long long)G * CH), TB, 0, stream>>>((unsigned*)gsum, 0u, (long long)G * CH);
    k_fill_u32<<<blocks1(G), TB, 0, stream>>>((unsigned*)cnt, 0u, G);
    k_pool_count<<<blocks1(N), TB, 0, stream>>>(batch, cnt, N);
    k_pool_sum<<<blocks1((long long)N * CH), TB, 0, stream>>>(hbuf, batch, gsum, N);
    k_build_comb<<<blocks1((long long)G * (CH + 2)), TB, 0, stream>>>(gsum, cnt, pegs_left, move_count, comb, G);

    float* out_p = (float*)d_out;          // p_logits [G,128]
    float* out_v = out_p + (long long)G * CH;  // v [G,1]
    k_mlp_relu<<<blocks1((long long)G * 128), TB, 0, stream>>>(comb, vW1, vb1, hidv, G, CH + 2, 128);
    k_mlp_relu<<<blocks1((long long)G * 128), TB, 0, stream>>>(comb, pW1, pb1, hidp, G, CH + 2, 128);
    k_head_policy<<<blocks1((long long)G * CH), TB, 0, stream>>>(hidp, pW2, pb2, out_p, G);
    k_head_value<<<blocks1(G), TB, 0, stream>>>(hidv, vW2, vb2, out_v, G);

    (void)n_in; (void)out_size; (void)ws_size;
}
